// Binary_CNN_22660247453833
// MI455X (gfx1250) — 8-bit matrix operand order repaired and checked by reading, compiled, not run on hardware
//
#include <hip/hip_runtime.h>
#include <stdint.h>

typedef __attribute__((ext_vector_type(8))) int v8i;

__global__ __launch_bounds__(256)
void pack_b2_kernel(const float* __restrict__ w2, int* __restrict__ b2p) {
  int idx = blockIdx.x * 256 + threadIdx.x;
  if (idx >= 32 * 18 * 32 * 8) return;
  int v    = idx & 7;
  int lane = (idx >> 3) & 31;
  int kc   = (idx >> 8) % 18;
  int nt   = idx / (8 * 32 * 18);
  int n    = nt * 16 + (lane & 15);
  int g    = lane >> 4;
  int d    = (v & 1) + (v >> 1) * 4 + g * 2;
  int K0   = kc * 64 + d * 4;
  unsigned dw = 0u;
  for (int j = 0; j < 4; ++j) {
    int K   = K0 + j;
    int tap = K >> 7;
    int c   = K & 127;
    float w = w2[n * 1152 + c * 9 + tap];
    unsigned by = (w >= 0.0f) ? 0x01u : 0xFFu;
    dw |= by << (8 * j);
  }
  b2p[idx] = (int)dw;
}

__global__ __launch_bounds__(256)
void pack_wf_kernel(const float* __restrict__ wf, int* __restrict__ wfb) {
  int idx = blockIdx.x * 256 + threadIdx.x;
  if (idx >= 128 * 32 * 8) return;
  int v    = idx & 7;
  int lane = (idx >> 3) & 31;
  int kc   = idx >> 8;
  int n    = lane & 15;
  int g    = lane >> 4;
  int d    = (v & 1) + (v >> 1) * 4 + g * 2;
  int K0   = kc * 64 + d * 4;
  unsigned dw = 0u;
  if (n < 10) {
    for (int j = 0; j < 4; ++j) {
      int K = K0 + j;
      int s = K >> 9;
      int c = K & 511;
      float w = wf[n * 8192 + c * 16 + s];
      unsigned by = (w >= 0.0f) ? 0x01u : 0xFFu;
      dw |= by << (8 * j);
    }
  }
  wfb[idx] = (int)dw;
}

__global__ __launch_bounds__(256)
void conv1_kernel(const float* __restrict__ x, const float* __restrict__ w1,
                  const float* __restrict__ b1, int8_t* __restrict__ a1) {
  __shared__ float xs[20][20];
  __shared__ float ws[3200];
  int t = threadIdx.x;
  int b = blockIdx.x;
  for (int i = t; i < 400; i += 256) (&xs[0][0])[i] = 0.0f;
  for (int i = t; i < 3200; i += 256) ws[i] = (w1[i] >= 0.0f) ? 1.0f : -1.0f;
  __syncthreads();
  { int r = t >> 4, c = t & 15; xs[r + 2][c + 2] = x[b * 256 + t]; }
  __syncthreads();

  for (int it = 0; it < 32; ++it) {
    int idx = it * 256 + t;
    int c   = idx & 127;
    int pos = idx >> 7;
    int py = pos >> 3, px = pos & 7;
    int oy = 2 * py, ox = 2 * px;
    float win[6][6];
#pragma unroll
    for (int r = 0; r < 6; ++r)
#pragma unroll
      for (int q = 0; q < 6; ++q) win[r][q] = xs[oy + r][ox + q];
    const float* wc = &ws[c * 25];
    float s00 = 0.f, s01 = 0.f, s10 = 0.f, s11 = 0.f;
#pragma unroll
    for (int ky = 0; ky < 5; ++ky)
#pragma unroll
      for (int kx = 0; kx < 5; ++kx) {
        float wv = wc[ky * 5 + kx];
        s00 = fmaf(wv, win[ky][kx],         s00);
        s01 = fmaf(wv, win[ky][kx + 1],     s01);
        s10 = fmaf(wv, win[ky + 1][kx],     s10);
        s11 = fmaf(wv, win[ky + 1][kx + 1], s11);
      }
    float m = fmaxf(fmaxf(s00, s01), fmaxf(s10, s11)) + b1[c];
    a1[b * 8192 + pos * 128 + c] = (m >= 0.0f) ? (int8_t)1 : (int8_t)-1;
  }
}

__global__ __launch_bounds__(256)
void conv2_kernel(const int* __restrict__ a1_4, const int* __restrict__ b2p,
                  const float* __restrict__ b2, int8_t* __restrict__ a2) {
  __shared__ int sA[16 * 9 * 32];
  int t  = threadIdx.x;
  int mt = blockIdx.x;
  int b  = mt >> 2, py = mt & 3;

  for (int j = t; j < 4608; j += 256) {
    int m   = j / 288;
    int rem = j - m * 288;
    int tap = rem >> 5;
    int c4  = rem & 31;
    int pos = py * 16 + m;
    int y = pos >> 3, xx = pos & 7;
    int ys = y + (tap / 3) - 1;
    int xs = xx + (tap % 3) - 1;
    int v = 0;
    if ((unsigned)ys < 8u && (unsigned)xs < 8u)
      v = a1_4[(b * 64 + ys * 8 + xs) * 32 + c4];
    sA[j] = v;
  }
  __syncthreads();

  int lane  = t & 31;
  int wv    = t >> 5;
  int g     = lane >> 4;
  int ml    = lane & 15;
  int ntile = blockIdx.y * 8 + wv;
  int n     = ntile * 16 + ml;

  v8i acc = {0, 0, 0, 0, 0, 0, 0, 0};
  const int* bp    = b2p + ((size_t)(ntile * 18) * 32 + lane) * 8;
  const int  abase = ml * 288;

#pragma unroll
  for (int kc = 0; kc < 18; ++kc) {
    int tap = kc >> 1;
    const int* ap = &sA[abase + tap * 32 + (kc & 1) * 16 + g * 2];
    v8i A;
    int2 p0 = *(const int2*)(ap + 0);
    int2 p1 = *(const int2*)(ap + 4);
    int2 p2 = *(const int2*)(ap + 8);
    int2 p3 = *(const int2*)(ap + 12);
    A[0] = p0.x; A[1] = p0.y; A[2] = p1.x; A[3] = p1.y;
    A[4] = p2.x; A[5] = p2.y; A[6] = p3.x; A[7] = p3.y;
    v8i Bv;
    int4 q0 = *(const int4*)(bp + kc * 256 + 0);
    int4 q1 = *(const int4*)(bp + kc * 256 + 4);
    Bv[0] = q0.x; Bv[1] = q0.y; Bv[2] = q0.z; Bv[3] = q0.w;
    Bv[4] = q1.x; Bv[5] = q1.y; Bv[6] = q1.z; Bv[7] = q1.w;
    acc = __builtin_amdgcn_wmma_i32_16x16x64_iu8(true, A, true, Bv, acc,
                                                 false, false);
  }

  float bn = b2[n];
  float f[8];
#pragma unroll
  for (int r = 0; r < 8; ++r) f[r] = (float)acc[r] + bn;
#pragma unroll
  for (int px = 0; px < 4; ++px) {
    float vo  = fmaxf(f[2 * px], f[2 * px + 1]);
    float ot  = __shfl_xor(vo, 16);
    float tot = fmaxf(vo, ot);
    if (g == 0)
      a2[((b * 4 + py) * 4 + px) * 512 + n] = (tot >= 0.0f) ? (int8_t)1
                                                            : (int8_t)-1;
  }
}

__global__ __launch_bounds__(32)
void fc_kernel(const int8_t* __restrict__ a2, const int* __restrict__ wfb,
               const float* __restrict__ bf, float* __restrict__ out) {
  int lane = threadIdx.x & 31;
  int mt   = blockIdx.x;
  int g    = lane >> 4;
  int ml   = lane & 15;
  const int8_t* arow = a2 + (size_t)(mt * 16 + ml) * 8192;

  v8i acc = {0, 0, 0, 0, 0, 0, 0, 0};
#pragma unroll 8
  for (int kc = 0; kc < 128; ++kc) {
    const int8_t* ab = arow + kc * 64 + g * 8;
    v8i A;
    int2 p0 = *(const int2*)(ab + 0);
    int2 p1 = *(const int2*)(ab + 16);
    int2 p2 = *(const int2*)(ab + 32);
    int2 p3 = *(const int2*)(ab + 48);
    A[0] = p0.x; A[1] = p0.y; A[2] = p1.x; A[3] = p1.y;
    A[4] = p2.x; A[5] = p2.y; A[6] = p3.x; A[7] = p3.y;
    const int* wp = wfb + ((size_t)kc * 32 + lane) * 8;
    __builtin_prefetch(wp + 512, 0, 1);
    v8i Bv;
    int4 q0 = *(const int4*)(wp + 0);
    int4 q1 = *(const int4*)(wp + 4);
    Bv[0] = q0.x; Bv[1] = q0.y; Bv[2] = q0.z; Bv[3] = q0.w;
    Bv[4] = q1.x; Bv[5] = q1.y; Bv[6] = q1.z; Bv[7] = q1.w;
    acc = __builtin_amdgcn_wmma_i32_16x16x64_iu8(true, A, true, Bv, acc,
                                                 false, false);
  }
  int nn = ml;
  float bias = (nn < 10) ? bf[nn] : 0.0f;
#pragma unroll
  for (int r = 0; r < 8; ++r) {
    int orow = mt * 16 + r + g * 8;
    if (nn < 10) out[orow * 10 + nn] = (float)acc[r] + bias;
  }
}

extern "C" void kernel_launch(void* const* d_in, const int* in_sizes, int n_in,
                              void* d_out, int out_size, void* d_ws,
                              size_t ws_size, hipStream_t stream) {
  const float* x  = (const float*)d_in[0];
  const float* w1 = (const float*)d_in[1];
  const float* b1 = (const float*)d_in[2];
  const float* w2 = (const float*)d_in[3];
  const float* b2 = (const float*)d_in[4];
  const float* wf = (const float*)d_in[5];
  const float* bf = (const float*)d_in[6];
  float* out = (float*)d_out;

  char* ws = (char*)d_ws;
  int8_t* a1  = (int8_t*)(ws);
  int8_t* a2  = (int8_t*)(ws + 33554432);
  int*    b2p = (int*)(ws + 67108864);
  int*    wfb = (int*)(ws + 67698688);

  pack_b2_kernel<<<576, 256, 0, stream>>>(w2, b2p);
  pack_wf_kernel<<<128, 256, 0, stream>>>(wf, wfb);
  conv1_kernel<<<4096, 256, 0, stream>>>(x, w1, b1, a1);
  conv2_kernel<<<dim3(16384, 4), 256, 0, stream>>>((const int*)a1, b2p, b2, a2);
  fc_kernel<<<256, 32, 0, stream>>>(a2, wfb, bf, out);
}
